// SpatialEncoder_9328668966996
// MI455X (gfx1250) — compile-verified, matrix operand order checked by reading, not run on hardware
//
#include <hip/hip_runtime.h>
#include <math.h>

#define D 128
#define NB_RED 256

typedef __attribute__((ext_vector_type(16))) _Float16 v16h;
typedef __attribute__((ext_vector_type(8)))  _Float16 v8h;
typedef __attribute__((ext_vector_type(4)))  _Float16 v4h;
typedef __attribute__((ext_vector_type(8)))  float    v8f;

// ---------------- device helpers ----------------

__device__ __forceinline__ float gelu_exact(float x) {
    return 0.5f * x * (1.0f + erff(x * 0.7071067811865475f));
}

__device__ __forceinline__ float wave_sum32(float v) {
#pragma unroll
    for (int off = 16; off > 0; off >>= 1) v += __shfl_xor(v, off, 32);
    return v;
}

// Build a 16-half fragment from two contiguous 16-byte runs (f16 already).
__device__ __forceinline__ v16h load_16h_pair(const _Float16* plo, const _Float16* phi) {
    v8h lo = *(const v8h*)plo;
    v8h hi = *(const v8h*)phi;
    v16h r;
#pragma unroll
    for (int e = 0; e < 8; ++e) { r[e] = lo[e]; r[e + 8] = hi[e]; }
    return r;
}

// C/D layout: VGPR j -> row (j + 8*(lane/16)), col (lane%16). Bias folded in.
__device__ __forceinline__ void store_c_bias(float* __restrict__ O, int row0, int col0, int ld,
                                             v8f c, int lane, const float* __restrict__ bias) {
    int n  = col0 + (lane & 15);
    int m0 = row0 + ((lane >> 4) << 3);
    float bv = bias[n];
#pragma unroll
    for (int j = 0; j < 8; ++j) O[(size_t)(m0 + j) * ld + n] = c[j] + bv;
}

// LayerNorm over a 128-wide row held as 4 regs/lane in a wave32
__device__ __forceinline__ void ln4(float v[4], const float* __restrict__ g,
                                    const float* __restrict__ b, int lane) {
    float s = v[0] + v[1] + v[2] + v[3];
    float mean = wave_sum32(s) * (1.0f / 128.0f);
    float q = 0.f;
#pragma unroll
    for (int j = 0; j < 4; ++j) { float d = v[j] - mean; q += d * d; }
    float var  = wave_sum32(q) * (1.0f / 128.0f);
    float rstd = rsqrtf(var + 1e-5f);
    int d0 = lane << 2;
#pragma unroll
    for (int j = 0; j < 4; ++j) v[j] = (v[j] - mean) * rstd * g[d0 + j] + b[d0 + j];
}

// ---------------- conversion / swizzle kernels ----------------

// generic f32 -> f16 (n multiple of 4)
__global__ void cvt4_kernel(const float* __restrict__ src, _Float16* __restrict__ dst, int n) {
    int i = (blockIdx.x * blockDim.x + threadIdx.x) << 2;
    if (i >= n) return;
    v4h o;
#pragma unroll
    for (int j = 0; j < 4; ++j) o[j] = (_Float16)src[i + j];
    *(v4h*)(dst + i) = o;
}

// W (K x 128 f32, row-major) -> fragment-swizzled f16:
// out[((kt*128 + col) * 32) + kl] = W[(kt*32+kl)*128 + col]
// so a lane's B fragment for (kt, col, half hf) is two contiguous b128 loads.
__global__ void swizzle_w_kernel(const float* __restrict__ W, _Float16* __restrict__ out, int K) {
    int idx = blockIdx.x * blockDim.x + threadIdx.x;   // over K*128
    if (idx >= K * D) return;
    int col = idx & (D - 1);
    int k   = idx >> 7;
    int kt  = k >> 5, kl = k & 31;
    out[(((kt << 7) + col) << 5) + kl] = (_Float16)W[idx];
}

// ---------------- WMMA GEMM (one wave per 16x16 tile, 8 waves = 8 col tiles) ----------------
// A in f16 row-major (split into A0 [KT0*32 wide] and A1 [(KT-KT0)*32 wide]);
// W pre-swizzled f16. out = A @ W + bias (f32).

template<int KT, int KT0>
__global__ void wmma_gemm_kernel(const _Float16* __restrict__ A0, int ldA0,
                                 const _Float16* __restrict__ A1, int ldA1,
                                 const _Float16* __restrict__ Wsw,
                                 const float* __restrict__ bias,
                                 float* __restrict__ out) {
    int lane = threadIdx.x & 31;
    int wv   = threadIdx.x >> 5;
    int row0 = blockIdx.x * 16;
    int col0 = wv * 16;
    int hf   = lane >> 4;
    int arow = row0 + (lane & 15);
    int col  = col0 + (lane & 15);
    v8f c = {};
#pragma unroll
    for (int kt = 0; kt < KT; ++kt) {
        const _Float16* Ap = (kt < KT0)
            ? A0 + (size_t)arow * ldA0 + (kt << 5)
            : A1 + (size_t)arow * ldA1 + ((kt - KT0) << 5);
        v16h a = load_16h_pair(Ap + (hf << 3), Ap + 16 + (hf << 3));
        const _Float16* Bp = Wsw + ((size_t)((kt << 7) + col) << 5);
        v16h b = load_16h_pair(Bp + (hf << 3), Bp + 16 + (hf << 3));
        c = __builtin_amdgcn_wmma_f32_16x16x32_f16(false, a, false, b, (short)0, c, false, false);
    }
    store_c_bias(out, row0, col0, D, c, lane, bias);
}

// ---------------- CSR build ----------------

__global__ void degree_int_kernel(const int* __restrict__ dst, int* __restrict__ deg, int E_) {
    int i = blockIdx.x * blockDim.x + threadIdx.x;
    if (i < E_) atomicAdd(&deg[dst[i]], 1);
}

__global__ void scan_block_kernel(const int* __restrict__ in, int* __restrict__ out,
                                  int* __restrict__ bsum, int n) {
    __shared__ int s[256];
    int i = blockIdx.x * 256 + threadIdx.x;
    int v = (i < n) ? in[i] : 0;
    s[threadIdx.x] = v; __syncthreads();
    for (int off = 1; off < 256; off <<= 1) {
        int t = (threadIdx.x >= off) ? s[threadIdx.x - off] : 0;
        __syncthreads();
        s[threadIdx.x] += t;
        __syncthreads();
    }
    if (i < n) out[i] = s[threadIdx.x] - v;                 // exclusive
    if (threadIdx.x == 255) bsum[blockIdx.x] = s[255];      // block total
}

__global__ void scan_bsum_kernel(int* __restrict__ bsum, int nb) {
    __shared__ int s[256];
    int v = (threadIdx.x < nb) ? bsum[threadIdx.x] : 0;
    s[threadIdx.x] = v; __syncthreads();
    for (int off = 1; off < 256; off <<= 1) {
        int t = (threadIdx.x >= off) ? s[threadIdx.x - off] : 0;
        __syncthreads();
        s[threadIdx.x] += t;
        __syncthreads();
    }
    if (threadIdx.x < nb) bsum[threadIdx.x] = s[threadIdx.x] - v;  // exclusive
}

__global__ void scan_add_kernel(int* __restrict__ out, const int* __restrict__ bsum, int n) {
    int i = blockIdx.x * 256 + threadIdx.x;
    if (i < n) out[i] += bsum[blockIdx.x];
}

__global__ void csr_fill_kernel(const int* __restrict__ src, const int* __restrict__ dst,
                                const int* __restrict__ rowstart, int* __restrict__ cursor,
                                int* __restrict__ csr_src, int E_) {
    int e = blockIdx.x * blockDim.x + threadIdx.x;
    if (e >= E_) return;
    int d = dst[e];
    int p = atomicAdd(&cursor[d], 1);
    csr_src[rowstart[d] + p] = src[e];
}

// ---------------- neighbor mean aggregation (no atomics) ----------------
// one wave per node; lane handles 4 dims; reads h16, writes agg16 = mean (f16)

__global__ void gather_kernel(const _Float16* __restrict__ h16, const int* __restrict__ csr_src,
                              const int* __restrict__ rowstart, const int* __restrict__ deg,
                              _Float16* __restrict__ agg16, int n) {
    int row  = (blockIdx.x * blockDim.x + threadIdx.x) >> 5;
    int lane = threadIdx.x & 31;
    if (row >= n) return;
    int beg = rowstart[row];
    int dg  = deg[row];
    int q   = lane << 2;
    float acc[4] = {0.f, 0.f, 0.f, 0.f};
    for (int j = beg; j < beg + dg; ++j) {
        int s = csr_src[j];
        v4h hv = *(const v4h*)(h16 + (size_t)s * D + q);
#pragma unroll
        for (int k = 0; k < 4; ++k) acc[k] += (float)hv[k];
    }
    float invd = 1.0f / (float)((dg > 0) ? dg : 1);
    v4h o;
#pragma unroll
    for (int k = 0; k < 4; ++k) o[k] = (_Float16)(acc[k] * invd);
    *(v4h*)(agg16 + (size_t)row * D + q) = o;
}

// ---------------- per-row epilogues (wave32 per 128-wide row) ----------------

__global__ void post_in_kernel(const float* __restrict__ tmp, float* __restrict__ h,
                               _Float16* __restrict__ h16,
                               const float* __restrict__ g, const float* __restrict__ b, int n) {
    int row  = (blockIdx.x * blockDim.x + threadIdx.x) >> 5;
    int lane = threadIdx.x & 31;
    if (row >= n) return;
    const float* r = tmp + (size_t)row * D + (lane << 2);
    float v[4] = {r[0], r[1], r[2], r[3]};
    ln4(v, g, b, lane);
#pragma unroll
    for (int j = 0; j < 4; ++j) v[j] = gelu_exact(v[j]);
    float* o = h + (size_t)row * D + (lane << 2);
    v4h o16;
#pragma unroll
    for (int j = 0; j < 4; ++j) { o[j] = v[j]; o16[j] = (_Float16)v[j]; }
    *(v4h*)(h16 + (size_t)row * D + (lane << 2)) = o16;
}

__global__ void layer_post_kernel(const float* __restrict__ tmp, float* __restrict__ h,
                                  _Float16* __restrict__ h16,
                                  const float* __restrict__ g1, const float* __restrict__ b1,
                                  const float* __restrict__ g2, const float* __restrict__ b2,
                                  int residual, int n) {
    int row  = (blockIdx.x * blockDim.x + threadIdx.x) >> 5;
    int lane = threadIdx.x & 31;
    if (row >= n) return;
    const float* r = tmp + (size_t)row * D + (lane << 2);
    float v[4];
#pragma unroll
    for (int j = 0; j < 4; ++j) v[j] = gelu_exact(r[j]);
    ln4(v, g1, b1, lane);
    ln4(v, g2, b2, lane);
    float* o = h + (size_t)row * D + (lane << 2);
    if (residual) {
        float old[4] = {o[0], o[1], o[2], o[3]};
#pragma unroll
        for (int j = 0; j < 4; ++j) v[j] = gelu_exact(v[j]) + old[j];
    }
    v4h o16;
#pragma unroll
    for (int j = 0; j < 4; ++j) { o[j] = v[j]; o16[j] = (_Float16)v[j]; }
    *(v4h*)(h16 + (size_t)row * D + (lane << 2)) = o16;
}

__global__ void gate_dot_kernel(const float* __restrict__ t, const float* __restrict__ w2,
                                const float* __restrict__ b2, float* __restrict__ logits, int n) {
    int row  = (blockIdx.x * blockDim.x + threadIdx.x) >> 5;
    int lane = threadIdx.x & 31;
    if (row >= n) return;
    const float* r = t + (size_t)row * D + (lane << 2);
    float acc = 0.f;
#pragma unroll
    for (int j = 0; j < 4; ++j) acc += tanhf(r[j]) * w2[(lane << 2) + j];
    acc = wave_sum32(acc);
    if (lane == 0) logits[row] = acc + b2[0];
}

// ---------------- softmax reductions ----------------

__global__ void reduce_max_partial(const float* __restrict__ x, int n, float* __restrict__ part) {
    __shared__ float s[256];
    float m = -3.4e38f;
    for (int i = blockIdx.x * blockDim.x + threadIdx.x; i < n; i += blockDim.x * gridDim.x)
        m = fmaxf(m, x[i]);
    s[threadIdx.x] = m; __syncthreads();
    for (int off = 128; off > 0; off >>= 1) {
        if (threadIdx.x < off) s[threadIdx.x] = fmaxf(s[threadIdx.x], s[threadIdx.x + off]);
        __syncthreads();
    }
    if (threadIdx.x == 0) part[blockIdx.x] = s[0];
}

__global__ void reduce_max_final(const float* __restrict__ part, float* __restrict__ scal) {
    __shared__ float s[256];
    s[threadIdx.x] = part[threadIdx.x]; __syncthreads();
    for (int off = 128; off > 0; off >>= 1) {
        if (threadIdx.x < off) s[threadIdx.x] = fmaxf(s[threadIdx.x], s[threadIdx.x + off]);
        __syncthreads();
    }
    if (threadIdx.x == 0) scal[0] = s[0];
}

__global__ void exp_partial(const float* __restrict__ logits, const float* __restrict__ scal,
                            float* __restrict__ gate, int n, float* __restrict__ part) {
    __shared__ float s[256];
    float m = scal[0];
    float acc = 0.f;
    for (int i = blockIdx.x * blockDim.x + threadIdx.x; i < n; i += blockDim.x * gridDim.x) {
        float e = expf(logits[i] - m);
        gate[i] = e;
        acc += e;
    }
    s[threadIdx.x] = acc; __syncthreads();
    for (int off = 128; off > 0; off >>= 1) {
        if (threadIdx.x < off) s[threadIdx.x] += s[threadIdx.x + off];
        __syncthreads();
    }
    if (threadIdx.x == 0) part[blockIdx.x] = s[0];
}

__global__ void reduce_sum_final(const float* __restrict__ part, float* __restrict__ scal) {
    __shared__ float s[256];
    s[threadIdx.x] = part[threadIdx.x]; __syncthreads();
    for (int off = 128; off > 0; off >>= 1) {
        if (threadIdx.x < off) s[threadIdx.x] += s[threadIdx.x + off];
        __syncthreads();
    }
    if (threadIdx.x == 0) scal[1] = s[0];
}

__global__ void normalize_gate(float* __restrict__ gate, const float* __restrict__ scal, int n) {
    int i = blockIdx.x * blockDim.x + threadIdx.x;
    if (i < n) gate[i] *= (1.0f / scal[1]);
}

// ---------------- pooling + output projection ----------------

__global__ void pool_kernel(const float* __restrict__ h, const float* __restrict__ gate,
                            const int* __restrict__ batch, float* __restrict__ pooled, int n) {
    __shared__ float acc[8 * D];
    for (int k = threadIdx.x; k < 8 * D; k += blockDim.x) acc[k] = 0.f;
    __syncthreads();
    int d   = threadIdx.x & (D - 1);
    int sub = threadIdx.x >> 7;
    for (int node = blockIdx.x * 2 + sub; node < n; node += gridDim.x * 2) {
        float v = h[(size_t)node * D + d] * gate[node];
        atomicAdd(&acc[(batch[node] << 7) + d], v);
    }
    __syncthreads();
    for (int k = threadIdx.x; k < 8 * D; k += blockDim.x) atomicAdd(&pooled[k], acc[k]);
}

__global__ void final_kernel(const float* __restrict__ pooled, const float* __restrict__ W,
                             const float* __restrict__ bias, const float* __restrict__ g,
                             const float* __restrict__ b, float* __restrict__ out) {
    __shared__ float s[8 * D];
    int t = threadIdx.x;                // 1024 threads
    int row = t >> 7, col = t & (D - 1);
    float acc = bias[col];
    for (int k = 0; k < D; ++k) acc += pooled[row * D + k] * W[k * D + col];
    s[t] = acc; __syncthreads();
    float mean = 0.f;
    for (int k = 0; k < D; ++k) mean += s[(row << 7) + k];
    mean *= (1.0f / 128.0f);
    float var = 0.f;
    for (int k = 0; k < D; ++k) { float d2 = s[(row << 7) + k] - mean; var += d2 * d2; }
    var *= (1.0f / 128.0f);
    float v = (acc - mean) * rsqrtf(var + 1e-5f) * g[col] + b[col];
    out[t] = gelu_exact(v);
}

// ---------------- launcher ----------------

extern "C" void kernel_launch(void* const* d_in, const int* in_sizes, int n_in,
                              void* d_out, int out_size, void* d_ws, size_t ws_size,
                              hipStream_t stream) {
    const float* x        = (const float*)d_in[0];
    const int*   ei       = (const int*)d_in[1];
    const int*   batch    = (const int*)d_in[2];
    const float* w_in     = (const float*)d_in[3];
    const float* b_in     = (const float*)d_in[4];
    const float* ln_in_g  = (const float*)d_in[5];
    const float* ln_in_b  = (const float*)d_in[6];
    const float* gnn_w    = (const float*)d_in[7];
    const float* gnn_b    = (const float*)d_in[8];
    const float* gnn_ln_g = (const float*)d_in[9];
    const float* gnn_ln_b = (const float*)d_in[10];
    const float* norm_g   = (const float*)d_in[11];
    const float* norm_b   = (const float*)d_in[12];
    const float* gate_w1  = (const float*)d_in[13];
    const float* gate_b1  = (const float*)d_in[14];
    const float* gate_w2  = (const float*)d_in[15];
    const float* gate_b2  = (const float*)d_in[16];
    const float* out_w    = (const float*)d_in[17];
    const float* out_b    = (const float*)d_in[18];
    const float* out_ln_g = (const float*)d_in[19];
    const float* out_ln_b = (const float*)d_in[20];

    const int N  = in_sizes[2];
    const int E  = in_sizes[1] / 2;
    const int L  = 3;
    const int Bg = 8;
    const int FIN = 64;
    const int* src = ei;
    const int* dst = ei + E;

    float* outf = (float*)d_out;
    float* emb  = outf;                 // 8 x 128
    float* h    = outf + Bg * D;        // N x 128 (f32, part of output)
    float* gate = h + (size_t)N * D;    // N

    // ---- workspace carve-up (256B-aligned chunks) ----
    char*  base = (char*)d_ws;
    size_t off  = 0;
    auto alloc = [&](size_t bytes) -> void* {
        void* p = base + off;
        off += (bytes + 255) & ~(size_t)255;
        return p;
    };
    float*    tmp      = (float*)   alloc((size_t)N * D * 4);      // GEMM f32 out
    _Float16* h16      = (_Float16*)alloc((size_t)N * D * 2);
    _Float16* agg16    = (_Float16*)alloc((size_t)N * D * 2);
    _Float16* x16      = (_Float16*)alloc((size_t)N * FIN * 2);
    _Float16* w_in_sw  = (_Float16*)alloc((size_t)FIN * D * 2);
    _Float16* gnn_sw   = (_Float16*)alloc((size_t)L * 2 * D * D * 2);
    _Float16* gate_sw  = (_Float16*)alloc((size_t)D * D * 2);
    int*      deg      = (int*)     alloc((size_t)N * 4);
    int*      rowstart = (int*)     alloc((size_t)N * 4);
    int*      cursor   = (int*)     alloc((size_t)N * 4);
    int*      csr_src  = (int*)     alloc((size_t)E * 4);
    int*      bsum     = (int*)     alloc(256 * 4);
    float*    logits   = (float*)   alloc((size_t)N * 4);
    float*    part     = (float*)   alloc(NB_RED * 4);
    float*    scal     = (float*)   alloc(2 * 4);
    float*    pooled   = (float*)   alloc((size_t)Bg * D * 4);

    const int rowTiles   = N / 16;                // 3125 (N % 16 == 0)
    const int rowBlocks  = (N + 7) / 8;           // wave-per-row kernels
    const int scanBlocks = (N + 255) / 256;       // 196 (<= 256)

    // ---- CSR build (once) ----
    hipMemsetAsync(deg, 0, (size_t)N * 4, stream);
    degree_int_kernel<<<(E + 255) / 256, 256, 0, stream>>>(dst, deg, E);
    scan_block_kernel<<<scanBlocks, 256, 0, stream>>>(deg, rowstart, bsum, N);
    scan_bsum_kernel<<<1, 256, 0, stream>>>(bsum, scanBlocks);
    scan_add_kernel<<<scanBlocks, 256, 0, stream>>>(rowstart, bsum, N);
    hipMemsetAsync(cursor, 0, (size_t)N * 4, stream);
    csr_fill_kernel<<<(E + 255) / 256, 256, 0, stream>>>(src, dst, rowstart, cursor, csr_src, E);

    // ---- weight prep: f16 + fragment swizzle ----
    swizzle_w_kernel<<<(FIN * D + 255) / 256, 256, 0, stream>>>(w_in, w_in_sw, FIN);
    for (int i = 0; i < L; ++i)
        swizzle_w_kernel<<<(2 * D * D + 255) / 256, 256, 0, stream>>>(
            gnn_w + (size_t)i * 2 * D * D, gnn_sw + (size_t)i * 2 * D * D, 2 * D);
    swizzle_w_kernel<<<(D * D + 255) / 256, 256, 0, stream>>>(gate_w1, gate_sw, D);

    // ---- input projection ----
    cvt4_kernel<<<((N * FIN / 4) + 255) / 256, 256, 0, stream>>>(x, x16, N * FIN);
    wmma_gemm_kernel<2, 2><<<rowTiles, 256, 0, stream>>>(x16, FIN, (const _Float16*)nullptr, 0,
                                                         w_in_sw, b_in, tmp);
    post_in_kernel<<<rowBlocks, 256, 0, stream>>>(tmp, h, h16, ln_in_g, ln_in_b, N);

    // ---- GNN layers ----
    for (int i = 0; i < L; ++i) {
        gather_kernel<<<rowBlocks, 256, 0, stream>>>(h16, csr_src, rowstart, deg, agg16, N);
        wmma_gemm_kernel<8, 4><<<rowTiles, 256, 0, stream>>>(
            h16, D, agg16, D, gnn_sw + (size_t)i * 2 * D * D, gnn_b + i * D, tmp);
        layer_post_kernel<<<rowBlocks, 256, 0, stream>>>(
            tmp, h, h16, gnn_ln_g + i * D, gnn_ln_b + i * D,
            norm_g + i * D, norm_b + i * D, (i < L - 1) ? 1 : 0, N);
    }

    // ---- attention gate ----
    wmma_gemm_kernel<4, 4><<<rowTiles, 256, 0, stream>>>(h16, D, (const _Float16*)nullptr, 0,
                                                         gate_sw, gate_b1, tmp);
    gate_dot_kernel<<<rowBlocks, 256, 0, stream>>>(tmp, gate_w2, gate_b2, logits, N);

    reduce_max_partial<<<NB_RED, 256, 0, stream>>>(logits, N, part);
    reduce_max_final<<<1, 256, 0, stream>>>(part, scal);
    exp_partial<<<NB_RED, 256, 0, stream>>>(logits, scal, gate, N, part);
    reduce_sum_final<<<1, 256, 0, stream>>>(part, scal);
    normalize_gate<<<(N + 255) / 256, 256, 0, stream>>>(gate, scal, N);

    hipMemsetAsync(pooled, 0, (size_t)(Bg * D) * 4, stream);
    pool_kernel<<<512, 256, 0, stream>>>(h, gate, batch, pooled, N);
    final_kernel<<<1, 1024, 0, stream>>>(pooled, out_w, out_b, out_ln_g, out_ln_b, emb);
}
